// VRNN_SINGLE_4294967296018
// MI455X (gfx1250) — matrix operand order repaired and checked by reading, not compiled, not run on hardware
//
#include <hip/hip_runtime.h>

#define DI __device__ __forceinline__

typedef __attribute__((ext_vector_type(16))) __bf16 v16bf;
typedef __attribute__((ext_vector_type(8)))  __bf16 bf8v;
typedef __attribute__((ext_vector_type(8)))  float  v8f;

namespace cfg {
constexpr int Y = 32, Z = 64, Hd = 256, R = 256, T = 200, Bb = 2048;
constexpr int BT = 16;           // batch rows per block
constexpr int NTHR = 256;        // 8 wave32
// bf16 weight offsets (elements) inside d_ws
constexpr size_t OFF_ENC_W1 = 0;
constexpr size_t OFF_ENC_W2 = OFF_ENC_W1 + 256ull*320;
constexpr size_t OFF_ENC_MW = OFF_ENC_W2 + 256ull*256;
constexpr size_t OFF_ENC_SW = OFF_ENC_MW + 64ull*256;
constexpr size_t OFF_PRI_W1 = OFF_ENC_SW + 64ull*256;
constexpr size_t OFF_PRI_W2 = OFF_PRI_W1 + 256ull*288;
constexpr size_t OFF_PRI_MW = OFF_PRI_W2 + 256ull*256;
constexpr size_t OFF_PRI_SW = OFF_PRI_MW + 64ull*256;
constexpr size_t OFF_DEC_W1 = OFF_PRI_SW + 64ull*256;
constexpr size_t OFF_DEC_W2 = OFF_DEC_W1 + 256ull*352;
constexpr size_t OFF_DEC_MW = OFF_DEC_W2 + 256ull*256;
constexpr size_t OFF_DEC_SW = OFF_DEC_MW + 32ull*256;
constexpr size_t OFF_GRU_WIH = OFF_DEC_SW + 32ull*256;
constexpr size_t OFF_GRU_WHH = OFF_GRU_WIH + 768ull*96;
constexpr size_t W_TOTAL     = OFF_GRU_WHH + 768ull*256;   // 794,624 bf16 = ~1.52 MB
// dynamic LDS layout (byte offsets), all 16B aligned
constexpr int S_XB  = 0;              // bf16 [16][32]
constexpr int S_YB  = S_XB  + 1024;   // bf16 [16][32]
constexpr int S_ZB  = S_YB  + 1024;   // bf16 [16][64]
constexpr int S_T1  = S_ZB  + 2048;   // bf16 [16][256]
constexpr int S_T2  = S_T1  + 8192;   // bf16 [16][256]
constexpr int S_HB0 = S_T2  + 8192;   // bf16 [16][256] (h, buf0)
constexpr int S_HB1 = S_HB0 + 8192;   // bf16 [16][256] (h, buf1)
constexpr int S_XF  = S_HB1 + 8192;   // f32  [16][32]
constexpr int S_E   = S_XF  + 2048;   // f32  [16][64]
constexpr int S_HF  = S_E   + 4096;   // f32  [16][256] (h fp32, single buf)
constexpr int S_EM  = S_HF  + 16384;  // f32  [16][64]
constexpr int S_ES  = S_EM  + 4096;
constexpr int S_PM  = S_ES  + 4096;
constexpr int S_PS  = S_PM  + 4096;
constexpr int S_DM  = S_PS  + 4096;   // f32  [16][32]
constexpr int S_DS  = S_DM  + 2048;
constexpr int S_RED = S_DS  + 2048;   // 64 B reduction scratch
constexpr int SMEM_BYTES = S_RED + 64;
}

// Segment descriptor: byte offset into dynamic LDS + row width in bf16 elements.
// (Offsets only — no LDS-derived pointers in aggregate initializers, which the
// linker cannot express as static initializers.)
struct Seg { int off; int width; };

// A fragment: 16x32 bf16, M = lane&15, two 8-elem runs at K = half*8 and 16+half*8
DI v16bf load_a_frag(const __bf16* seg, int segW, int kin, int m, int half) {
  const __bf16* p = seg + m * segW + kin + half * 8;
  bf8v lo = *(const bf8v*)p;
  bf8v hi = *(const bf8v*)(p + 16);
  v16bf a;
#pragma unroll
  for (int i = 0; i < 8; ++i) { a[i] = lo[i]; a[i + 8] = hi[i]; }
  return a;
}

// B fragment: 32x16 bf16 (K x N), N = lane&15, two 8-elem runs at K = half*8 and 16+half*8 (in W[n][k])
DI v16bf load_b_frag(const __bf16* wrow, int k, int half) {
  const __bf16* p = wrow + k + half * 8;
  bf8v lo = *(const bf8v*)p;
  bf8v hi = *(const bf8v*)(p + 16);
  v16bf b;
#pragma unroll
  for (int i = 0; i < 8; ++i) { b[i] = lo[i]; b[i + 8] = hi[i]; }
  return b;
}

template <int NS>
DI v8f mm_tile(const char* sm, const Seg (&segs)[NS], const __bf16* W, int K,
               int n0, int lane) {
  const int q = lane & 15, half = lane >> 4;
  v8f acc = {};
  const __bf16* wrow = W + (size_t)(n0 + q) * (size_t)K;
  int kbase = 0;
#pragma unroll
  for (int s = 0; s < NS; ++s) {
    const __bf16* base = (const __bf16*)(sm + segs[s].off);
    const int w = segs[s].width;
    for (int kt = 0; kt < w; kt += 32) {
      v16bf a = load_a_frag(base, w, kt, q, half);
      v16bf b = load_b_frag(wrow, kbase + kt, half);
      acc = __builtin_amdgcn_wmma_f32_16x16x32_bf16(false, a, false, b,
                                                    (short)0, acc, false, false);
    }
    kbase += w;
  }
  return acc;
}

DI float softplusf(float x) { return x > 20.f ? x : log1pf(__expf(x)); }
DI float sigmoidf(float x)  { return 1.f / (1.f + __expf(-x)); }

// D layout: lane holds D[r + half*8][n0 + (lane&15)], r = 0..7
DI void store_relu_bf16(v8f acc, const float* bias, int n0, __bf16* out, int outW, int lane) {
  const int n = n0 + (lane & 15), half = lane >> 4;
  const float bv = bias[n];
#pragma unroll
  for (int r = 0; r < 8; ++r) {
    float v = acc[r] + bv;
    v = v > 0.f ? v : 0.f;
    out[(r + half * 8) * outW + n] = (__bf16)v;
  }
}

DI void store_head_f32(v8f acc, const float* bias, int n0, float* out, int outW, int lane, bool sp) {
  const int n = n0 + (lane & 15), half = lane >> 4;
  const float bv = bias[n];
#pragma unroll
  for (int r = 0; r < 8; ++r) {
    float v = acc[r] + bv;
    if (sp) v = softplusf(v);
    out[(r + half * 8) * outW + n] = v;
  }
}

struct VrnnParams {
  const float* states; const float* eps;
  const __bf16* W;
  const float *enc_b1, *enc_b2, *enc_mb, *enc_sb;
  const float *pri_b1, *pri_b2, *pri_mb, *pri_sb;
  const float *dec_b1, *dec_b2, *dec_mb, *dec_sb;
  const float *gru_bih, *gru_bhh;
  float* out;
};

__global__ void vrnn_init_out(float* out) {
  if (threadIdx.x < 2) out[threadIdx.x] = 0.f;
}

__global__ void vrnn_cvt_bf16(const float* src, __bf16* dst, int n) {
  for (int i = blockIdx.x * blockDim.x + threadIdx.x; i < n; i += gridDim.x * blockDim.x)
    dst[i] = (__bf16)src[i];
}

__global__ __launch_bounds__(cfg::NTHR)
void vrnn_persistent(VrnnParams p) {
  using namespace cfg;
  extern __shared__ __align__(16) char smem[];
  char*   sm   = smem;
  __bf16* sXb  = (__bf16*)(sm + S_XB);
  __bf16* sYb  = (__bf16*)(sm + S_YB);
  __bf16* sZb  = (__bf16*)(sm + S_ZB);
  __bf16* sT1  = (__bf16*)(sm + S_T1);
  __bf16* sHb0 = (__bf16*)(sm + S_HB0);
  __bf16* sHb1 = (__bf16*)(sm + S_HB1);
  float*  sXf  = (float*)(sm + S_XF);
  float*  sE   = (float*)(sm + S_E);
  float*  sHf  = (float*)(sm + S_HF);
  float*  sEM  = (float*)(sm + S_EM);
  float*  sES  = (float*)(sm + S_ES);
  float*  sPM  = (float*)(sm + S_PM);
  float*  sPS  = (float*)(sm + S_PS);
  float*  sDM  = (float*)(sm + S_DM);
  float*  sDS  = (float*)(sm + S_DS);
  float*  sRed = (float*)(sm + S_RED);

  const int tid  = threadIdx.x;
  const int lane = tid & 31;
  const int wave = tid >> 5;
  const int rowBase = blockIdx.x * BT;

  // h0 = 0
  for (int i = tid; i < BT * R; i += NTHR) { sHf[i] = 0.f; sHb0[i] = (__bf16)0.f; }
  float klAcc = 0.f, nllAcc = 0.f;
  __syncthreads();

  int cur = 0;
#pragma clang loop unroll(disable)
  for (int t = 0; t < T - 1; ++t) {
    const int hOffC = (cur == 0) ? S_HB0 : S_HB1;   // current h (bf16) LDS offset
    __bf16* hbN = (cur == 0) ? sHb1 : sHb0;         // next h buffer

    // ---- phase 1: stage y = states[t], x = states[t+1], eps[t] into LDS ----
    const float* yrow = p.states + (size_t)t * Bb * Y + (size_t)rowBase * Y;
    const float* xrow = yrow + (size_t)Bb * Y;
    for (int i = tid; i < BT * Y; i += NTHR) {
      float yv = yrow[i]; sYb[i] = (__bf16)yv;
      float xv = xrow[i]; sXb[i] = (__bf16)xv; sXf[i] = xv;
    }
    const float* erow = p.eps + (size_t)t * Bb * Z + (size_t)rowBase * Z;
    for (int i = tid; i < BT * Z; i += NTHR) sE[i] = erow[i];
    if (t + 2 < T) {                       // prefetch next step (L2/L0 warm)
      const float* pf = p.states + (size_t)(t + 2) * Bb * Y + (size_t)rowBase * Y;
      __builtin_prefetch(pf + tid * 2, 0, 3);
      const float* pe = p.eps + (size_t)(t + 1) * Bb * Z + (size_t)rowBase * Z;
      __builtin_prefetch(pe + tid * 4, 0, 3);
    }
    __syncthreads();

    // ---- encoder: relu(relu([x,y,h] W1^T + b1) W2^T + b2) ----
    {
      Seg s3[3] = { { S_XB, 32 }, { S_YB, 32 }, { hOffC, 256 } };
#pragma unroll
      for (int i = 0; i < 2; ++i) {
        int n0 = (wave * 2 + i) * 16;
        v8f acc = mm_tile<3>(sm, s3, p.W + OFF_ENC_W1, 320, n0, lane);
        store_relu_bf16(acc, p.enc_b1, n0, sT1, 256, lane);
      }
    }
    __syncthreads();
    {
      Seg s1[1] = { { S_T1, 256 } };
#pragma unroll
      for (int i = 0; i < 2; ++i) {
        int n0 = (wave * 2 + i) * 16;
        v8f acc = mm_tile<1>(sm, s1, p.W + OFF_ENC_W2, 256, n0, lane);
        store_relu_bf16(acc, p.enc_b2, n0, (__bf16*)(sm + S_T2), 256, lane);
      }
    }
    __syncthreads();
    {
      Seg s1[1] = { { S_T2, 256 } };
      if (wave < 4) {
        int n0 = wave * 16;
        v8f acc = mm_tile<1>(sm, s1, p.W + OFF_ENC_MW, 256, n0, lane);
        store_head_f32(acc, p.enc_mb, n0, sEM, 64, lane, false);
      } else {
        int n0 = (wave - 4) * 16;
        v8f acc = mm_tile<1>(sm, s1, p.W + OFF_ENC_SW, 256, n0, lane);
        store_head_f32(acc, p.enc_sb, n0, sES, 64, lane, true);
      }
    }
    __syncthreads();

    // ---- prior: [y,h] ----
    {
      Seg s2[2] = { { S_YB, 32 }, { hOffC, 256 } };
#pragma unroll
      for (int i = 0; i < 2; ++i) {
        int n0 = (wave * 2 + i) * 16;
        v8f acc = mm_tile<2>(sm, s2, p.W + OFF_PRI_W1, 288, n0, lane);
        store_relu_bf16(acc, p.pri_b1, n0, sT1, 256, lane);
      }
    }
    __syncthreads();
    {
      Seg s1[1] = { { S_T1, 256 } };
#pragma unroll
      for (int i = 0; i < 2; ++i) {
        int n0 = (wave * 2 + i) * 16;
        v8f acc = mm_tile<1>(sm, s1, p.W + OFF_PRI_W2, 256, n0, lane);
        store_relu_bf16(acc, p.pri_b2, n0, (__bf16*)(sm + S_T2), 256, lane);
      }
    }
    __syncthreads();
    {
      Seg s1[1] = { { S_T2, 256 } };
      if (wave < 4) {
        int n0 = wave * 16;
        v8f acc = mm_tile<1>(sm, s1, p.W + OFF_PRI_MW, 256, n0, lane);
        store_head_f32(acc, p.pri_mb, n0, sPM, 64, lane, false);
      } else {
        int n0 = (wave - 4) * 16;
        v8f acc = mm_tile<1>(sm, s1, p.W + OFF_PRI_SW, 256, n0, lane);
        store_head_f32(acc, p.pri_sb, n0, sPS, 64, lane, true);
      }
    }
    __syncthreads();

    // ---- z = em + es*eps ; KL accumulation ----
    for (int i = tid; i < BT * Z; i += NTHR) {
      float em = sEM[i], es = sES[i], pm = sPM[i], ps = sPS[i];
      float zv = em + es * sE[i];
      sZb[i] = (__bf16)zv;
      float d = em - pm;
      klAcc += 0.5f * (2.f * __logf(ps / es) + (es * es + d * d) / (ps * ps) - 1.f);
    }
    __syncthreads();

    // ---- decoder: [y,z,h] ----
    {
      Seg s3[3] = { { S_YB, 32 }, { S_ZB, 64 }, { hOffC, 256 } };
#pragma unroll
      for (int i = 0; i < 2; ++i) {
        int n0 = (wave * 2 + i) * 16;
        v8f acc = mm_tile<3>(sm, s3, p.W + OFF_DEC_W1, 352, n0, lane);
        store_relu_bf16(acc, p.dec_b1, n0, sT1, 256, lane);
      }
    }
    __syncthreads();
    {
      Seg s1[1] = { { S_T1, 256 } };
#pragma unroll
      for (int i = 0; i < 2; ++i) {
        int n0 = (wave * 2 + i) * 16;
        v8f acc = mm_tile<1>(sm, s1, p.W + OFF_DEC_W2, 256, n0, lane);
        store_relu_bf16(acc, p.dec_b2, n0, (__bf16*)(sm + S_T2), 256, lane);
      }
    }
    __syncthreads();
    {
      Seg s1[1] = { { S_T2, 256 } };
      if (wave < 2) {
        int n0 = wave * 16;
        v8f acc = mm_tile<1>(sm, s1, p.W + OFF_DEC_MW, 256, n0, lane);
        store_head_f32(acc, p.dec_mb, n0, sDM, 32, lane, false);
      } else if (wave < 4) {
        int n0 = (wave - 2) * 16;
        v8f acc = mm_tile<1>(sm, s1, p.W + OFF_DEC_SW, 256, n0, lane);
        store_head_f32(acc, p.dec_sb, n0, sDS, 32, lane, true);
      }
    }
    __syncthreads();

    // ---- NLL accumulation ----
    for (int i = tid; i < BT * Y; i += NTHR) {
      float ds = sDS[i], dm = sDM[i], xv = sXf[i];
      float dd = xv - dm;
      nllAcc += __logf(ds) + dd * dd / (2.f * ds * ds) + 0.9189385332046727f;
    }
    // no barrier needed: GRU reads sXb/sZb/h(cur) and writes only h(next)/own sHf cells

    // ---- GRU: gi = [x,z] Wih^T + bih ; gh = h Whh^T + bhh ----
    {
      Seg gis[2] = { { S_XB, 32 }, { S_ZB, 64 } };
      Seg ghs[1] = { { hOffC, 256 } };
#pragma unroll
      for (int i = 0; i < 2; ++i) {
        const int jt = wave * 2 + i, j0 = jt * 16;
        v8f ir  = mm_tile<2>(sm, gis, p.W + OFF_GRU_WIH, 96, j0, lane);
        v8f iz  = mm_tile<2>(sm, gis, p.W + OFF_GRU_WIH, 96, 256 + j0, lane);
        v8f inn = mm_tile<2>(sm, gis, p.W + OFF_GRU_WIH, 96, 512 + j0, lane);
        v8f hr  = mm_tile<1>(sm, ghs, p.W + OFF_GRU_WHH, 256, j0, lane);
        v8f hz  = mm_tile<1>(sm, ghs, p.W + OFF_GRU_WHH, 256, 256 + j0, lane);
        v8f hn  = mm_tile<1>(sm, ghs, p.W + OFF_GRU_WHH, 256, 512 + j0, lane);
        const int n = j0 + (lane & 15), half = lane >> 4;
        const float bir = p.gru_bih[n], biz = p.gru_bih[256 + n], bin = p.gru_bih[512 + n];
        const float bhr = p.gru_bhh[n], bhz = p.gru_bhh[256 + n], bhn = p.gru_bhh[512 + n];
#pragma unroll
        for (int r = 0; r < 8; ++r) {
          float rg = sigmoidf((ir[r] + bir) + (hr[r] + bhr));
          float ug = sigmoidf((iz[r] + biz) + (hz[r] + bhz));
          float ng = tanhf((inn[r] + bin) + rg * (hn[r] + bhn));
          const int idx = (r + half * 8) * R + n;           // this thread owns idx
          float hold = sHf[idx];
          float hnew = (1.f - ug) * ng + ug * hold;
          sHf[idx] = hnew;
          hbN[idx] = (__bf16)hnew;
        }
      }
    }
    __syncthreads();
    cur ^= 1;
  }

  // ---- final reduction: wave shuffle -> LDS -> global atomic ----
#pragma unroll
  for (int off = 16; off > 0; off >>= 1) {
    klAcc  += __shfl_down(klAcc, off);
    nllAcc += __shfl_down(nllAcc, off);
  }
  if (lane == 0) { sRed[wave] = klAcc; sRed[8 + wave] = nllAcc; }
  __syncthreads();
  if (tid == 0) {
    float k = 0.f, nl = 0.f;
#pragma unroll
    for (int i = 0; i < 8; ++i) { k += sRed[i]; nl += sRed[8 + i]; }
    atomicAdd(&p.out[0], k);
    atomicAdd(&p.out[1], nl);
  }
}

extern "C" void kernel_launch(void* const* d_in, const int* in_sizes, int n_in,
                              void* d_out, int out_size, void* d_ws, size_t ws_size,
                              hipStream_t stream) {
  using namespace cfg;
  (void)in_sizes; (void)n_in; (void)out_size; (void)ws_size;

  __bf16* Wb = (__bf16*)d_ws;

  // fp32 -> bf16 weight staging into d_ws (L2-resident thereafter)
  struct Cvt { int idx; size_t off; int n; };
  const Cvt items[14] = {
    {  2, OFF_ENC_W1, 256 * 320 }, {  4, OFF_ENC_W2, 256 * 256 },
    {  6, OFF_ENC_MW,  64 * 256 }, {  8, OFF_ENC_SW,  64 * 256 },
    { 10, OFF_PRI_W1, 256 * 288 }, { 12, OFF_PRI_W2, 256 * 256 },
    { 14, OFF_PRI_MW,  64 * 256 }, { 16, OFF_PRI_SW,  64 * 256 },
    { 18, OFF_DEC_W1, 256 * 352 }, { 20, OFF_DEC_W2, 256 * 256 },
    { 22, OFF_DEC_MW,  32 * 256 }, { 24, OFF_DEC_SW,  32 * 256 },
    { 26, OFF_GRU_WIH, 768 * 96 }, { 27, OFF_GRU_WHH, 768 * 256 },
  };
  for (int i = 0; i < 14; ++i) {
    int blocks = (items[i].n + 255) / 256;
    vrnn_cvt_bf16<<<blocks, 256, 0, stream>>>((const float*)d_in[items[i].idx],
                                              Wb + items[i].off, items[i].n);
  }

  vrnn_init_out<<<1, 32, 0, stream>>>((float*)d_out);

  VrnnParams p;
  p.states = (const float*)d_in[0];
  p.eps    = (const float*)d_in[1];
  p.W      = Wb;
  p.enc_b1 = (const float*)d_in[3];  p.enc_b2 = (const float*)d_in[5];
  p.enc_mb = (const float*)d_in[7];  p.enc_sb = (const float*)d_in[9];
  p.pri_b1 = (const float*)d_in[11]; p.pri_b2 = (const float*)d_in[13];
  p.pri_mb = (const float*)d_in[15]; p.pri_sb = (const float*)d_in[17];
  p.dec_b1 = (const float*)d_in[19]; p.dec_b2 = (const float*)d_in[21];
  p.dec_mb = (const float*)d_in[23]; p.dec_sb = (const float*)d_in[25];
  p.gru_bih = (const float*)d_in[28]; p.gru_bhh = (const float*)d_in[29];
  p.out = (float*)d_out;

  vrnn_persistent<<<Bb / BT, NTHR, SMEM_BYTES, stream>>>(p);
}
